// GCN_FFN_1425929142799
// MI455X (gfx1250) — matrix operand order repaired and checked by reading, not compiled, not run on hardware
//
#include <hip/hip_runtime.h>
#include <hip/hip_bf16.h>

typedef __bf16 bf16_t;
typedef bf16_t v16bf __attribute__((ext_vector_type(16)));
typedef bf16_t v8bf  __attribute__((ext_vector_type(8)));
typedef bf16_t v4bf  __attribute__((ext_vector_type(4)));
typedef float  v8f   __attribute__((ext_vector_type(8)));
typedef float  v4f   __attribute__((ext_vector_type(4)));

#define N_NODES 512
#define F_DIM   256
#define T_DIM   64
#define B_DIM   4
#define M_WG    128   // query rows per workgroup (8 waves x 16)
#define KC      64    // key/value node tile
#define QSTR    264   // padded LDS row stride for 256-wide rows (bf16 elems)
#define VSTR    72    // padded LDS row stride for 64-wide rows
#define SCALE   0.0625f   // 1/sqrt(256)

// ---- WMMA A-operand load: lane (g,m) holds row m, K = 16g + e (contiguous 16) ----
__device__ __forceinline__ v16bf lds_loadA(const bf16_t* row, int koff, int g) {
  v8bf lo = *(const v8bf*)(row + koff + 16*g);
  v8bf hi = *(const v8bf*)(row + koff + 16*g + 8);
  return __builtin_shufflevector(lo, hi, 0,1,2,3,4,5,6,7,8,9,10,11,12,13,14,15);
}
// ---- WMMA B-operand load: lane (g,n) holds col n, K = 16g + e (contiguous 16) ----
__device__ __forceinline__ v16bf lds_loadB(const bf16_t* row, int koff, int g) {
  v8bf lo = *(const v8bf*)(row + koff + 16*g);
  v8bf hi = *(const v8bf*)(row + koff + 16*g + 8);
  return __builtin_shufflevector(lo, hi, 0,1,2,3,4,5,6,7,8,9,10,11,12,13,14,15);
}
__device__ __forceinline__ v8f wmma_bf16(v16bf a, v16bf b, v8f c) {
  return __builtin_amdgcn_wmma_f32_16x16x32_bf16(false, a, false, b, (short)0, c, false, false);
}

__global__ __launch_bounds__(256, 1) void gcn_attn_fused(
    const float* __restrict__ x, const float* __restrict__ adj,
    const float* __restrict__ theta, float* __restrict__ out)
{
  // 128x264 Q | 64x264 Knat | 256x72 Vtr | 8 waves x 16x72 P  = 78336 bf16 = 153 KB
  __shared__ __align__(16) bf16_t smem[78336];
  bf16_t* Qs   = smem;           // staged Q rows (later reused to stage h)
  bf16_t* Knat = smem + 33792;   // K tile, node-major
  bf16_t* Vtr  = smem + 50688;   // V tile, feature-major (transposed)
  bf16_t* Pb   = smem + 69120;   // per-wave probability scratch

  const int bt = blockIdx.x >> 2;      // folded batch (B*T)
  const int rt = blockIdx.x & 3;       // row tile within N
  const int bb = bt >> 6;
  const int tt = bt & 63;
  const int row_base = rt * M_WG;

  const float* xb = x + ((size_t)bb * N_NODES * T_DIM + tt) * F_DIM; // row n at + n*T*F

  const int tid  = threadIdx.x;
  const int wave = tid >> 5, lane = tid & 31;
  const int g    = lane >> 4, hl = lane & 15;

  // ---- stage Q tile (fp32 -> bf16), resident for whole kernel ----
  {
    const int rsub = tid >> 6;
    const int f4   = (tid & 63) * 4;
    #pragma unroll 4
    for (int r0 = 0; r0 < M_WG; r0 += 4) {
      const int r = r0 + rsub;
      const v4f xv = *(const v4f*)(xb + (size_t)(row_base + r) * (T_DIM * F_DIM) + f4);
      v4bf q = { (bf16_t)xv.x, (bf16_t)xv.y, (bf16_t)xv.z, (bf16_t)xv.w };
      *(v4bf*)(Qs + r * QSTR + f4) = q;
    }
  }

  float m_run[8], l_run[8];
  v8f acc[16];
  #pragma unroll
  for (int j = 0; j < 8; ++j) { m_run[j] = -1e30f; l_run[j] = 0.f; }
  #pragma unroll
  for (int ft = 0; ft < 16; ++ft)
    #pragma unroll
    for (int j = 0; j < 8; ++j) acc[ft][j] = 0.f;

  const bf16_t* Qw = Qs + wave * 16 * QSTR;
  bf16_t*       Pw = Pb + wave * 16 * VSTR;

  for (int cb = 0; cb < N_NODES; cb += KC) {
    __syncthreads();   // previous tile's LDS reads done before overwrite
    // ---- cooperative K/V tile load: fp32 -> bf16, natural + transposed copies ----
    {
      const int rsub = tid >> 6;
      const int f4   = (tid & 63) * 4;
      #pragma unroll 4
      for (int r0 = 0; r0 < KC; r0 += 4) {
        const int r = r0 + rsub;
        const v4f xv = *(const v4f*)(xb + (size_t)(cb + r) * (T_DIM * F_DIM) + f4);
        bf16_t e0 = (bf16_t)xv.x, e1 = (bf16_t)xv.y, e2 = (bf16_t)xv.z, e3 = (bf16_t)xv.w;
        v4bf kv = { e0, e1, e2, e3 };
        *(v4bf*)(Knat + r * QSTR + f4) = kv;
        Vtr[(f4 + 0) * VSTR + r] = e0;
        Vtr[(f4 + 1) * VSTR + r] = e1;
        Vtr[(f4 + 2) * VSTR + r] = e2;
        Vtr[(f4 + 3) * VSTR + r] = e3;
      }
    }
    __syncthreads();

    // ---- S = (Q @ K^T) * scale : 16 rows x 64 cols per wave ----
    float sc[4][8];
    #pragma unroll
    for (int ct = 0; ct < 4; ++ct) {
      v8f s = {0.f,0.f,0.f,0.f,0.f,0.f,0.f,0.f};
      const bf16_t* Arow = Qw + hl * QSTR;
      const bf16_t* Brow = Knat + (16 * ct + hl) * QSTR;
      #pragma unroll
      for (int kc = 0; kc < 8; ++kc)
        s = wmma_bf16(lds_loadA(Arow, 32 * kc, g), lds_loadB(Brow, 32 * kc, g), s);
      #pragma unroll
      for (int j = 0; j < 8; ++j) sc[ct][j] = s[j] * SCALE;
    }

    // ---- online softmax; numerator gets adj weighting, denominator does not ----
    #pragma unroll
    for (int j = 0; j < 8; ++j) {
      float tmax = fmaxf(fmaxf(sc[0][j], sc[1][j]), fmaxf(sc[2][j], sc[3][j]));
      tmax = fmaxf(tmax, __shfl_xor(tmax, 1));
      tmax = fmaxf(tmax, __shfl_xor(tmax, 2));
      tmax = fmaxf(tmax, __shfl_xor(tmax, 4));
      tmax = fmaxf(tmax, __shfl_xor(tmax, 8));
      const float mnew = fmaxf(m_run[j], tmax);
      const float fs   = __expf(m_run[j] - mnew);
      m_run[j] = mnew;
      float rsum = 0.f;
      const size_t arow = (size_t)(row_base + wave * 16 + j + 8 * g) * N_NODES + cb + hl;
      #pragma unroll
      for (int ct = 0; ct < 4; ++ct) {
        const float e = __expf(sc[ct][j] - mnew);
        rsum += e;
        const float p = e * adj[arow + 16 * ct];
        Pw[(j + 8 * g) * VSTR + 16 * ct + hl] = (bf16_t)p;  // D-layout -> row-major P
      }
      rsum += __shfl_xor(rsum, 1);
      rsum += __shfl_xor(rsum, 2);
      rsum += __shfl_xor(rsum, 4);
      rsum += __shfl_xor(rsum, 8);
      l_run[j] = l_run[j] * fs + rsum;
      #pragma unroll
      for (int ft = 0; ft < 16; ++ft) acc[ft][j] *= fs;
    }
    __asm volatile("s_wait_dscnt 0" ::: "memory");  // wave-internal P visibility

    // ---- acc += P @ V (K = 64 as two K=32 WMMAs) ----
    const bf16_t* Prow = Pw + hl * VSTR;
    #pragma unroll
    for (int ft = 0; ft < 16; ++ft) {
      const bf16_t* Brow = Vtr + (16 * ft + hl) * VSTR;
      #pragma unroll
      for (int kb = 0; kb < 2; ++kb)
        acc[ft] = wmma_bf16(lds_loadA(Prow, 32 * kb, g), lds_loadB(Brow, 32 * kb, g), acc[ft]);
    }
  }

  // ---- h = acc * scale / l ; stage into our own Q rows (Q no longer needed) ----
  #pragma unroll
  for (int j = 0; j < 8; ++j) {
    const float inv = SCALE / l_run[j];
    #pragma unroll
    for (int ft = 0; ft < 16; ++ft)
      Qs[(wave * 16 + j + 8 * g) * QSTR + 16 * ft + hl] = (bf16_t)(acc[ft][j] * inv);
  }
  __asm volatile("s_wait_dscnt 0" ::: "memory");

  // ---- y = relu(h @ W^T); W row-major [o][k] is exactly B-layout, read from L2 ----
  const bf16_t* Hrow  = Qw + hl * QSTR;
  const size_t  obase = ((size_t)(bb * N_NODES + row_base + wave * 16) * T_DIM + tt) * F_DIM;
  for (int ot = 0; ot < 16; ++ot) {
    v8f y = {0.f,0.f,0.f,0.f,0.f,0.f,0.f,0.f};
    const float* Wrow = theta + (size_t)(16 * ot + hl) * F_DIM;
    #pragma unroll
    for (int kc = 0; kc < 8; ++kc) {
      const v4f w0 = *(const v4f*)(Wrow + 32 * kc + 16 * g + 0);
      const v4f w1 = *(const v4f*)(Wrow + 32 * kc + 16 * g + 4);
      const v4f w2 = *(const v4f*)(Wrow + 32 * kc + 16 * g + 8);
      const v4f w3 = *(const v4f*)(Wrow + 32 * kc + 16 * g + 12);
      v16bf bv;
      bv[0]=(bf16_t)w0.x;  bv[1]=(bf16_t)w0.y;  bv[2]=(bf16_t)w0.z;  bv[3]=(bf16_t)w0.w;
      bv[4]=(bf16_t)w1.x;  bv[5]=(bf16_t)w1.y;  bv[6]=(bf16_t)w1.z;  bv[7]=(bf16_t)w1.w;
      bv[8]=(bf16_t)w2.x;  bv[9]=(bf16_t)w2.y;  bv[10]=(bf16_t)w2.z; bv[11]=(bf16_t)w2.w;
      bv[12]=(bf16_t)w3.x; bv[13]=(bf16_t)w3.y; bv[14]=(bf16_t)w3.z; bv[15]=(bf16_t)w3.w;
      y = wmma_bf16(lds_loadA(Hrow, 32 * kc, g), bv, y);
    }
    #pragma unroll
    for (int j = 0; j < 8; ++j)
      out[obase + (size_t)(j + 8 * g) * (T_DIM * F_DIM) + 16 * ot + hl] = fmaxf(y[j], 0.f);
  }
}

extern "C" void kernel_launch(void* const* d_in, const int* in_sizes, int n_in,
                              void* d_out, int out_size, void* d_ws, size_t ws_size,
                              hipStream_t stream) {
  (void)in_sizes; (void)n_in; (void)d_ws; (void)ws_size; (void)out_size;
  const float* x     = (const float*)d_in[0];
  const float* adj   = (const float*)d_in[1];
  const float* theta = (const float*)d_in[2];
  float*       out   = (float*)d_out;
  const int blocks = B_DIM * T_DIM * (N_NODES / M_WG);  // 256 batches x 4 row tiles = 1024
  gcn_attn_fused<<<blocks, 256, 0, stream>>>(x, adj, theta, out);
}
